// MinimalLlamaAttention_29343216566540
// MI455X (gfx1250) — matrix operand order repaired and checked by reading, not compiled, not run on hardware
//
#include <hip/hip_runtime.h>

typedef __bf16 bf16_t;
typedef __bf16 bf16x8 __attribute__((ext_vector_type(8)));
typedef __bf16 v16bf  __attribute__((ext_vector_type(16)));
typedef float  v8f    __attribute__((ext_vector_type(8)));

#define B_   2
#define S_   2048
#define D_   2048
#define H_   32
#define KV_  8
#define DH_  64
#define MTOT (B_*S_)   // 4096 rows

#define LP 40          // LDS row pitch in elements (32 + 8 pad -> conflict-free, 16B aligned)

// round-to-nearest-even f32 -> bf16
__device__ inline bf16_t f2bf(float f) {
    unsigned u = __builtin_bit_cast(unsigned, f);
    unsigned r = (u + 0x7FFFu + ((u >> 16) & 1u)) >> 16;
    return __builtin_bit_cast(bf16_t, (unsigned short)r);
}

__device__ inline v16bf combine8(bf16x8 lo, bf16x8 hi) {
    v16bf r;
#pragma unroll
    for (int i = 0; i < 8; ++i) { r[i] = lo[i]; r[i + 8] = hi[i]; }
    return r;
}

__global__ void cast_f32_bf16(const float* __restrict__ in, bf16_t* __restrict__ out, int n) {
    int i = blockIdx.x * blockDim.x + threadIdx.x;
    if (i < n) out[i] = f2bf(in[i]);
}

// C[M,N] = A[M,K] (bf16) @ W[N,K]^T (bf16), f32 out.
// Block = 256 threads = 8 waves -> 128x128 C tile. K stepped by 32 with
// double-buffered LDS staging of A(128x32) and B(128x32). Each wave: 16 M-rows,
// all 128 N-cols (8 accumulators), 8 WMMAs per K-step. All fragments are
// loaded before the WMMA burst so the compiler can issue one DS clause, one
// wait, then 8 back-to-back WMMAs.
__global__ __launch_bounds__(256) void gemm_bf16_wmma(
    const bf16_t* __restrict__ A, const bf16_t* __restrict__ W,
    float* __restrict__ C, int M, int N, int K) {
    __shared__ __align__(16) bf16_t sA[2][128 * LP];
    __shared__ __align__(16) bf16_t sB[2][128 * LP];

    const int tilesN = N >> 7;
    const int tM = blockIdx.x / tilesN;
    const int tN = blockIdx.x % tilesN;

    const int t    = threadIdx.x;
    const int lane = t & 31;
    const int wv   = t >> 5;
    const int n15  = lane & 15;
    const int hh   = lane >> 4;
    const int mbase = wv * 16;

    // cooperative staging: 512 16B-chunks per tile; thread t handles rows r0 and r0+64
    const int r0 = t >> 2;
    const int c0 = (t & 3) * 8;
    const bf16_t* gA0 = A + (size_t)(tM * 128 + r0) * K;
    const bf16_t* gA1 = gA0 + (size_t)64 * K;
    const bf16_t* gB0 = W + (size_t)(tN * 128 + r0) * K;
    const bf16_t* gB1 = gB0 + (size_t)64 * K;

    v8f acc[8];
#pragma unroll
    for (int i = 0; i < 8; ++i) acc[i] = (v8f){};

    // preload k-step 0 into buffer 0
    bf16x8 ra0 = *(const bf16x8*)(gA0 + c0);
    bf16x8 ra1 = *(const bf16x8*)(gA1 + c0);
    bf16x8 rb0 = *(const bf16x8*)(gB0 + c0);
    bf16x8 rb1 = *(const bf16x8*)(gB1 + c0);
    *(bf16x8*)(&sA[0][r0 * LP + c0])        = ra0;
    *(bf16x8*)(&sA[0][(r0 + 64) * LP + c0]) = ra1;
    *(bf16x8*)(&sB[0][r0 * LP + c0])        = rb0;
    *(bf16x8*)(&sB[0][(r0 + 64) * LP + c0]) = rb1;

    int buf = 0;
    for (int kb = 0; kb < K; kb += 32) {
        const bool more = (kb + 32) < K;
        if (more) {
            ra0 = *(const bf16x8*)(gA0 + kb + 32 + c0);
            ra1 = *(const bf16x8*)(gA1 + kb + 32 + c0);
            rb0 = *(const bf16x8*)(gB0 + kb + 32 + c0);
            rb1 = *(const bf16x8*)(gB1 + kb + 32 + c0);
        }
        __syncthreads();   // buffer `buf` fully staged; prior reads of buf^1 done

        // pull ALL fragments for this K-step into registers first
        const bf16_t* ar = &sA[buf][(mbase + n15) * LP];
        v16bf a = combine8(*(const bf16x8*)(ar + hh * 16),
                           *(const bf16x8*)(ar + hh * 16 + 8));
        v16bf bfr[8];
#pragma unroll
        for (int nt = 0; nt < 8; ++nt) {
            const bf16_t* br = &sB[buf][(nt * 16 + n15) * LP + hh * 16];
            bfr[nt] = combine8(*(const bf16x8*)(br), *(const bf16x8*)(br + 8));
        }
        if (more) {
            *(bf16x8*)(&sA[buf ^ 1][r0 * LP + c0])        = ra0;
            *(bf16x8*)(&sA[buf ^ 1][(r0 + 64) * LP + c0]) = ra1;
            *(bf16x8*)(&sB[buf ^ 1][r0 * LP + c0])        = rb0;
            *(bf16x8*)(&sB[buf ^ 1][(r0 + 64) * LP + c0]) = rb1;
        }
        // WMMA burst
#pragma unroll
        for (int nt = 0; nt < 8; ++nt) {
            acc[nt] = __builtin_amdgcn_wmma_f32_16x16x32_bf16(false, a, false, bfr[nt],
                                                              (short)0, acc[nt], false, false);
        }
        buf ^= 1;
    }
#pragma unroll
    for (int nt = 0; nt < 8; ++nt) {
#pragma unroll
        for (int r = 0; r < 8; ++r) {
            int row = tM * 128 + mbase + r + 8 * hh;
            int col = tN * 128 + nt * 16 + n15;
            C[(size_t)row * N + col] = acc[nt][r];
        }
    }
}

// RoPE on f32 projections, emit bf16 [B, heads, S, DH]
__global__ void rope_to_bf16(const float* __restrict__ src, bf16_t* __restrict__ dst,
                             int heads, int total) {
    int i = blockIdx.x * blockDim.x + threadIdx.x;
    if (i >= total) return;
    int cols = heads * DH_;
    int m = i / cols, c = i % cols;
    int hh = c / DH_, d = c % DH_;
    int b = m / S_, s = m % S_;
    float ang = (float)s * __powf(10000.0f, -(float)(d & 31) / 32.0f);
    float cs = __cosf(ang), sn = __sinf(ang);
    float x0 = src[(size_t)m * cols + c];
    float xr = (d < 32) ? -src[(size_t)m * cols + hh * DH_ + d + 32]
                        :  src[(size_t)m * cols + hh * DH_ + d - 32];
    dst[(((size_t)b * heads + hh) * S_ + s) * DH_ + d] = f2bf(x0 * cs + xr * sn);
}

// V cast + transpose: f32 [B*S, KV*DH] -> bf16 [B, KV, DH, S]
__global__ void v_transpose_bf16(const float* __restrict__ src, bf16_t* __restrict__ dst, int total) {
    int i = blockIdx.x * blockDim.x + threadIdx.x;
    if (i >= total) return;
    int m = i / (KV_ * DH_), c = i % (KV_ * DH_);
    int kv = c / DH_, d = c % DH_;
    int b = m / S_, s = m % S_;
    dst[(((size_t)b * KV_ + kv) * DH_ + d) * S_ + s] = f2bf(src[i]);
}

// Causal GQA flash attention. One wave = 16 query rows of one (b,h).
// Q: [B,H,S,DH] bf16 (RoPEd), K: [B,KV,S,DH] bf16 (RoPEd), Vt: [B,KV,DH,S] bf16.
// O: [B*S, H*DH] bf16.
__global__ __launch_bounds__(128) void attn_fwd_wmma(
    const bf16_t* __restrict__ Q, const bf16_t* __restrict__ Kc,
    const bf16_t* __restrict__ Vt, bf16_t* __restrict__ O) {
    __shared__ __align__(16) bf16_t plds[4 * 16 * 32];
    const int lane = threadIdx.x & 31;
    const int w    = threadIdx.x >> 5;
    const int wid  = blockIdx.x * 4 + w;
    const int qt = wid & 127;          // S/16 = 128
    const int h  = (wid >> 7) & 31;
    const int b  = wid >> 12;
    const int kv = h >> 2;             // GQA: rep = 4
    const int n15 = lane & 15, hh = lane >> 4;

    const bf16_t* qbase = Q  + ((size_t)b * H_  + h ) * S_ * DH_;
    const bf16_t* kbase = Kc + ((size_t)b * KV_ + kv) * S_ * DH_;
    const bf16_t* vbase = Vt + ((size_t)b * KV_ + kv) * DH_ * S_;
    bf16_t* pl = plds + w * 16 * 32;

    // Q fragments (A-layout, M = n15, two K=32 halves of DH=64)
    const bf16_t* qrowp = qbase + (size_t)(qt * 16 + n15) * DH_;
    v16bf qf0 = combine8(*(const bf16x8*)(qrowp + hh * 16),
                         *(const bf16x8*)(qrowp + 8 + hh * 16));
    v16bf qf1 = combine8(*(const bf16x8*)(qrowp + 32 + hh * 16),
                         *(const bf16x8*)(qrowp + 40 + hh * 16));

    float mrow[8], lrow[8];
    v8f o0 = {}, o1 = {}, o2 = {}, o3 = {};
#pragma unroll
    for (int r = 0; r < 8; ++r) { mrow[r] = -1e30f; lrow[r] = 0.0f; }

    const int nkt = (qt * 16 + 47) >> 5;   // 32-key steps covering causal extent
    for (int kt = 0; kt < nkt; ++kt) {
        const int t0 = kt * 32;
        // prefetch next key/value tile into cache (global_prefetch_b8)
        if (t0 + 32 + n15 < S_) {
            __builtin_prefetch(kbase + (size_t)(t0 + 32 + n15) * DH_ + hh * 16, 0, 3);
            __builtin_prefetch(vbase + (size_t)(n15) * S_ + t0 + 32 + hh * 16, 0, 3);
        }
        v8f s0 = {}, s1 = {};
        {
            const bf16_t* kr = kbase + (size_t)(t0 + n15) * DH_ + hh * 16;
            v16bf b0 = combine8(*(const bf16x8*)(kr),      *(const bf16x8*)(kr + 8));
            v16bf b1 = combine8(*(const bf16x8*)(kr + 32), *(const bf16x8*)(kr + 40));
            s0 = __builtin_amdgcn_wmma_f32_16x16x32_bf16(false, qf0, false, b0, (short)0, s0, false, false);
            s0 = __builtin_amdgcn_wmma_f32_16x16x32_bf16(false, qf1, false, b1, (short)0, s0, false, false);
        }
        {
            const bf16_t* kr = kbase + (size_t)(t0 + 16 + n15) * DH_ + hh * 16;
            v16bf b0 = combine8(*(const bf16x8*)(kr),      *(const bf16x8*)(kr + 8));
            v16bf b1 = combine8(*(const bf16x8*)(kr + 32), *(const bf16x8*)(kr + 40));
            s1 = __builtin_amdgcn_wmma_f32_16x16x32_bf16(false, qf0, false, b0, (short)0, s1, false, false);
            s1 = __builtin_amdgcn_wmma_f32_16x16x32_bf16(false, qf1, false, b1, (short)0, s1, false, false);
        }
        // online softmax; row r+8*hh lives in VGPR r of this lane's half
#pragma unroll
        for (int r = 0; r < 8; ++r) {
            int qrow = qt * 16 + r + 8 * hh;
            int c0 = t0 + n15, c1 = t0 + 16 + n15;
            float a0 = (c0 <= qrow) ? s0[r] * 0.125f : -1e30f;
            float a1 = (c1 <= qrow) ? s1[r] * 0.125f : -1e30f;
            float tm = fmaxf(a0, a1);
            tm = fmaxf(tm, __shfl_xor(tm, 1, 32));
            tm = fmaxf(tm, __shfl_xor(tm, 2, 32));
            tm = fmaxf(tm, __shfl_xor(tm, 4, 32));
            tm = fmaxf(tm, __shfl_xor(tm, 8, 32));
            float nm   = fmaxf(mrow[r], tm);
            float corr = __expf(mrow[r] - nm);
            float p0 = (c0 <= qrow) ? __expf(a0 - nm) : 0.0f;
            float p1 = (c1 <= qrow) ? __expf(a1 - nm) : 0.0f;
            float rs = p0 + p1;
            rs += __shfl_xor(rs, 1, 32);
            rs += __shfl_xor(rs, 2, 32);
            rs += __shfl_xor(rs, 4, 32);
            rs += __shfl_xor(rs, 8, 32);
            lrow[r] = lrow[r] * corr + rs;
            mrow[r] = nm;
            o0[r] *= corr; o1[r] *= corr; o2[r] *= corr; o3[r] *= corr;
            pl[(r + 8 * hh) * 32 + n15]      = f2bf(p0);
            pl[(r + 8 * hh) * 32 + 16 + n15] = f2bf(p1);
        }
        asm volatile("s_wait_dscnt 0" ::: "memory");  // wave-level LDS visibility
        // P fragment in A-layout (M = n15, K = 32 keys) from LDS
        const bf16_t* pr = pl + n15 * 32;
        v16bf pf = combine8(*(const bf16x8*)(pr + hh * 16),
                            *(const bf16x8*)(pr + 8 + hh * 16));
        // P (16x32) x V (32x16 per 16-col tile of DH)
#pragma unroll
        for (int nt = 0; nt < 4; ++nt) {
            const bf16_t* vr = vbase + (size_t)(nt * 16 + n15) * S_ + t0 + hh * 16;
            v16bf vf = combine8(*(const bf16x8*)(vr), *(const bf16x8*)(vr + 8));
            v8f& oacc = (nt == 0) ? o0 : (nt == 1) ? o1 : (nt == 2) ? o2 : o3;
            oacc = __builtin_amdgcn_wmma_f32_16x16x32_bf16(false, pf, false, vf, (short)0, oacc, false, false);
        }
    }
    // normalize and store bf16 attn output [B*S, H*DH]
#pragma unroll
    for (int nt = 0; nt < 4; ++nt) {
        v8f oacc = (nt == 0) ? o0 : (nt == 1) ? o1 : (nt == 2) ? o2 : o3;
#pragma unroll
        for (int r = 0; r < 8; ++r) {
            int qrow = qt * 16 + r + 8 * hh;
            size_t row = (size_t)b * S_ + qrow;
            O[row * (H_ * DH_) + h * DH_ + nt * 16 + n15] = f2bf(oacc[r] / lrow[r]);
        }
    }
}

extern "C" void kernel_launch(void* const* d_in, const int* in_sizes, int n_in,
                              void* d_out, int out_size, void* d_ws, size_t ws_size,
                              hipStream_t stream) {
    (void)in_sizes; (void)n_in; (void)out_size; (void)ws_size;
    const float* x  = (const float*)d_in[0];
    const float* Wq = (const float*)d_in[1];
    const float* Wk = (const float*)d_in[2];
    const float* Wv = (const float*)d_in[3];
    const float* Wo = (const float*)d_in[4];
    float* out = (float*)d_out;

    char* ws = (char*)d_ws;
    size_t off = 0;
    auto alloc = [&](size_t bytes) -> char* {
        char* p = ws + off;
        off += (bytes + 255) & ~(size_t)255;
        return p;
    };

    bf16_t* xb  = (bf16_t*)alloc((size_t)MTOT * D_ * 2);
    bf16_t* wqb = (bf16_t*)alloc((size_t)H_ * DH_ * D_ * 2);
    bf16_t* wkb = (bf16_t*)alloc((size_t)KV_ * DH_ * D_ * 2);
    bf16_t* wvb = (bf16_t*)alloc((size_t)KV_ * DH_ * D_ * 2);
    bf16_t* wob = (bf16_t*)alloc((size_t)D_ * H_ * DH_ * 2);
    float*  qf  = (float*) alloc((size_t)MTOT * H_ * DH_ * 4);
    float*  kf  = (float*) alloc((size_t)MTOT * KV_ * DH_ * 4);
    float*  vf  = (float*) alloc((size_t)MTOT * KV_ * DH_ * 4);
    bf16_t* Qb  = (bf16_t*)alloc((size_t)B_ * H_ * S_ * DH_ * 2);
    bf16_t* Kb  = (bf16_t*)alloc((size_t)B_ * KV_ * S_ * DH_ * 2);
    bf16_t* Vt  = (bf16_t*)alloc((size_t)B_ * KV_ * S_ * DH_ * 2);
    bf16_t* attnb = xb;  // x's bf16 copy is dead after the QKV GEMMs; reuse it

    auto cdiv = [](int a, int b) { return (a + b - 1) / b; };

    cast_f32_bf16<<<cdiv(MTOT * D_, 256), 256, 0, stream>>>(x,  xb,  MTOT * D_);
    cast_f32_bf16<<<cdiv(H_ * DH_ * D_, 256), 256, 0, stream>>>(Wq, wqb, H_ * DH_ * D_);
    cast_f32_bf16<<<cdiv(KV_ * DH_ * D_, 256), 256, 0, stream>>>(Wk, wkb, KV_ * DH_ * D_);
    cast_f32_bf16<<<cdiv(KV_ * DH_ * D_, 256), 256, 0, stream>>>(Wv, wvb, KV_ * DH_ * D_);
    cast_f32_bf16<<<cdiv(D_ * H_ * DH_, 256), 256, 0, stream>>>(Wo, wob, D_ * H_ * DH_);

    // QKV projections: grid = (M/128)*(N/128) blocks of 256 threads
    gemm_bf16_wmma<<<(MTOT / 128) * (2048 / 128), 256, 0, stream>>>(xb, wqb, qf, MTOT, 2048, D_);
    gemm_bf16_wmma<<<(MTOT / 128) * (512  / 128), 256, 0, stream>>>(xb, wkb, kf, MTOT, 512,  D_);
    gemm_bf16_wmma<<<(MTOT / 128) * (512  / 128), 256, 0, stream>>>(xb, wvb, vf, MTOT, 512,  D_);

    rope_to_bf16<<<cdiv(MTOT * H_ * DH_, 256), 256, 0, stream>>>(qf, Qb, H_,  MTOT * H_ * DH_);
    rope_to_bf16<<<cdiv(MTOT * KV_ * DH_, 256), 256, 0, stream>>>(kf, Kb, KV_, MTOT * KV_ * DH_);
    v_transpose_bf16<<<cdiv(MTOT * KV_ * DH_, 256), 256, 0, stream>>>(vf, Vt, MTOT * KV_ * DH_);

    // one wave per (b,h,16-query-row tile); 4 waves per 128-thread block
    attn_fwd_wmma<<<(B_ * H_ * (S_ / 16)) / 4, 128, 0, stream>>>(Qb, Kb, Vt, attnb);

    // output projection straight into d_out (f32)
    gemm_bf16_wmma<<<(MTOT / 128) * (2048 / 128), 256, 0, stream>>>(attnb, wob, out, MTOT, 2048, D_);
}
